// My_Net_43215960932471
// MI455X (gfx1250) — matrix operand order repaired and checked by reading, not compiled, not run on hardware
//
#include <hip/hip_runtime.h>
#include <stdint.h>
#include <stddef.h>

#define EPS_F 1e-5f

// ---------------------------------------------------------------------------
// WMMA types (CDNA5 / gfx1250, wave32)
// ---------------------------------------------------------------------------
typedef __attribute__((ext_vector_type(16))) __bf16 v16bf;
typedef __attribute__((ext_vector_type(8)))  float  v8f;

union Frag16 { v16bf v; uint32_t u[8]; };

// ---------------------------------------------------------------------------
// Weight packing: fold BN scale into weights, emit fused bias, zero-pad K and
// pad Cout to a multiple of 64 (zero rows) so the conv kernel can stage the
// A-tile with unguarded async LDS loads.
// ---------------------------------------------------------------------------
__global__ void pack_conv_weights(const float* __restrict__ w,
                                  const float* __restrict__ g,
                                  const float* __restrict__ beta,
                                  const float* __restrict__ mean,
                                  const float* __restrict__ var,
                                  const float* __restrict__ cbias,
                                  __bf16* __restrict__ wpack,
                                  float* __restrict__ biasf,
                                  int Co, int CoPad, int K, int Kpad)
{
    int i = blockIdx.x * blockDim.x + threadIdx.x;
    if (i >= CoPad * Kpad) return;
    int co = i / Kpad, k = i - co * Kpad;
    if (co >= Co) { wpack[i] = (__bf16)0.f; return; }
    float s = g[co] * rsqrtf(var[co] + EPS_F);
    float val = (k < K) ? w[(size_t)co * K + k] * s : 0.f;
    wpack[i] = (__bf16)val;
    if (k == 0)
        biasf[co] = ((cbias ? cbias[co] : 0.f) - mean[co]) * s + beta[co];
}

// ---------------------------------------------------------------------------
// Implicit-GEMM conv + folded BN (+optional residual) + optional ReLU.
// Workgroup tile: 64 Cout x 64 pixels (8 waves); each wave owns two 16x16
// C/D tiles -> 2 x v_wmma_f32_16x16x32_bf16 per 32-wide K chunk with the A
// fragment reused. Double-buffered LDS (one barrier per chunk); weight tile
// staged memory->LDS via global_load_async_to_lds_b128 (ASYNCcnt).
// R,S compile-time so the im2col decomposition uses constant divisors.
// ---------------------------------------------------------------------------
template <int R, int Sk>
__global__ __launch_bounds__(256)
void conv_bn_relu_wmma(const float* __restrict__ in,   // [B][Ci][IH][IW]
                       const __bf16* __restrict__ wp,  // [CoPad][Kpad] packed
                       const float* __restrict__ bias, // [Co] fused
                       const float* __restrict__ res,  // [B][Co][OH][OW] or null
                       float* __restrict__ out,        // [B][CoTotal][OH][OW]
                       int Ci, int IH, int IW,
                       int Co, int OH, int OW,
                       int stride, int pad,
                       int K, int Kpad, int CoTotal, int doRelu)
{
    constexpr int RS = R * Sk;
    __shared__ __bf16 lA[2][64][32];   // Cout x K chunk, double buffered (8 KB)
    __shared__ __bf16 lB[2][64][32];   // pixel x K chunk, double buffered (8 KB)
    const int b      = blockIdx.z;
    const int coBase = blockIdx.y * 64;
    const int pxBase = blockIdx.x * 64;
    const int tid    = threadIdx.x;
    const int lane   = tid & 31;
    const int wv     = tid >> 5;
    const int wr     = wv >> 1;     // 0..3 -> Cout sub-tile
    const int wc     = wv & 1;      // 0..1 -> 32-pixel half
    const int OHW    = OH * OW;
    const float* inB = in + (size_t)b * Ci * IH * IW;

    // A staging: thread -> weight row (tid>>2), 16-byte chunk (tid&3). One
    // async b128 per thread covers the whole 64x32 bf16 tile.
    const int arow = tid >> 2, a16 = tid & 3;
    const uint64_t gbase =
        (uint64_t)(uintptr_t)(wp + (size_t)(coBase + arow) * Kpad) + (uint64_t)a16 * 16;

    // B staging: thread -> pixel row (tid>>2), 8 bf16 along K ((tid&3)*8)
    const int prow = tid >> 2;
    const int kcol = (tid & 3) << 3;
    const int px_s = pxBase + prow;
    const bool pok = px_s < OHW;
    const int oy_s = pok ? px_s / OW : 0;
    const int ox_s = pok ? px_s - oy_s * OW : 0;
    const int iy0  = oy_s * stride - pad;
    const int ix0  = ox_s * stride - pad;

    v8f acc0 = {}, acc1 = {};

    auto stageA = [&](int k0, int buf) {
        uint32_t laddr = (uint32_t)(uintptr_t)(&lA[buf][arow][0]) + (uint32_t)(a16 * 16);
        uint64_t gaddr = gbase + (uint64_t)k0 * 2;
        asm volatile("global_load_async_to_lds_b128 %0, %1, off"
                     :: "v"(laddr), "v"(gaddr) : "memory");
    };
    auto stageB = [&](int k0, int buf) {
        #pragma unroll
        for (int e = 0; e < 8; ++e) {
            int k = k0 + kcol + e;
            float val = 0.f;
            if (pok && k < K) {
                int ci  = k / RS;          // compile-time divisor
                int rem = k - ci * RS;
                int ry  = rem / Sk;        // compile-time divisor
                int sx  = rem - ry * Sk;
                int iy  = iy0 + ry;
                int ix  = ix0 + sx;
                if (iy >= 0 && iy < IH && ix >= 0 && ix < IW)
                    val = inB[((size_t)ci * IH + iy) * IW + ix];
            }
            lB[buf][prow][kcol + e] = (__bf16)val;
        }
    };

    stageA(0, 0);
    stageB(0, 0);
    asm volatile("s_wait_asynccnt 0" ::: "memory");
    __syncthreads();

    const int m  = lane & 15;
    const int hi = lane >> 4;

    int buf = 0;
    for (int k0 = 0; k0 < K; k0 += 32, buf ^= 1) {
        const bool more = (k0 + 32) < K;
        if (more) { stageA(k0 + 32, buf ^ 1); stageB(k0 + 32, buf ^ 1); }

        // fragments: contiguous k order (k = 16*hi + s), same for A and B
        Frag16 af, bf0, bf1;
        const uint32_t* arow32 = (const uint32_t*)&lA[buf][wr * 16 + m][0];
        const uint32_t* b0r    = (const uint32_t*)&lB[buf][wc * 32 + m][0];
        const uint32_t* b1r    = (const uint32_t*)&lB[buf][wc * 32 + 16 + m][0];
        #pragma unroll
        for (int e = 0; e < 8; ++e) {
            af.u[e]     = arow32[hi * 8 + e];       // K 0-15 / 16-31
        }
        #pragma unroll
        for (int e = 0; e < 8; ++e) {
            bf0.u[e] = b0r[hi * 8 + e];             // K 0-15 / 16-31
            bf1.u[e] = b1r[hi * 8 + e];
        }

        acc0 = __builtin_amdgcn_wmma_f32_16x16x32_bf16(
                   false, af.v, false, bf0.v, (short)0, acc0, false, false);
        acc1 = __builtin_amdgcn_wmma_f32_16x16x32_bf16(
                   false, af.v, false, bf1.v, (short)0, acc1, false, false);

        if (more) asm volatile("s_wait_asynccnt 0" ::: "memory");
        __syncthreads();
    }

    // epilogue: D layout VGPR e -> M=e (lanes 0-15), M=e+8 (lanes 16-31)
    const int co0 = coBase + wr * 16;
    #pragma unroll
    for (int t = 0; t < 2; ++t) {
        const v8f& a = t ? acc1 : acc0;
        const int px = pxBase + wc * 32 + t * 16 + (lane & 15);
        if (px < OHW) {
            const int oy = px / OW, ox = px - oy * OW;
            #pragma unroll
            for (int e = 0; e < 8; ++e) {
                int co = co0 + e + ((lane >> 4) << 3);
                if (co < Co) {
                    float val = a[e] + bias[co];
                    if (res) val += res[(((size_t)b * Co + co) * OH + oy) * OW + ox];
                    if (doRelu) val = fmaxf(val, 0.f);
                    out[(((size_t)b * CoTotal + co) * OH + oy) * OW + ox] = val;
                }
            }
        }
    }
}

// ---------------------------------------------------------------------------
// Deconv (lhs_dilation=2, pad (2,3), 5x5) + BN + ReLU, direct gather.
// ---------------------------------------------------------------------------
__global__ void deconv_bn_relu(const float* __restrict__ in,
                               const float* __restrict__ w,
                               const float* __restrict__ cb,
                               const float* __restrict__ g,
                               const float* __restrict__ be,
                               const float* __restrict__ mn,
                               const float* __restrict__ vr,
                               float* __restrict__ out,
                               int Bn, int IH, int IW,
                               int OH, int OW, int CoTotal)
{
    const int C = 64;
    size_t i = blockIdx.x * (size_t)blockDim.x + threadIdx.x;
    size_t total = (size_t)Bn * C * OH * OW;
    if (i >= total) return;
    int ox = (int)(i % OW);
    int oy = (int)((i / OW) % OH);
    int co = (int)((i / ((size_t)OW * OH)) % C);
    int b  = (int)(i / ((size_t)OW * OH * C));
    float acc = 0.f;
    for (int ky = 0; ky < 5; ++ky) {
        int t = oy + ky - 2;
        if (t < 0 || t >= 2 * IH - 1 || (t & 1)) continue;
        int iy = t >> 1;
        for (int kx = 0; kx < 5; ++kx) {
            int u = ox + kx - 2;
            if (u < 0 || u >= 2 * IW - 1 || (u & 1)) continue;
            int ix = u >> 1;
            const float* wc = w + co * 25 + (4 - ky) * 5 + (4 - kx);
            const float* ib = in + (size_t)b * C * IH * IW + (size_t)iy * IW + ix;
            #pragma unroll 4
            for (int ci = 0; ci < C; ++ci)
                acc += wc[(size_t)ci * C * 25] * ib[(size_t)ci * IH * IW];
        }
    }
    float s = g[co] * rsqrtf(vr[co] + EPS_F);
    float val = (acc + cb[co] - mn[co]) * s + be[co];
    out[(((size_t)b * CoTotal + co) * OH + oy) * OW + ox] = fmaxf(val, 0.f);
}

// ---------------------------------------------------------------------------
// Superpixel statistics + sampling machinery
// ---------------------------------------------------------------------------
__global__ void zero_f32(float* p, size_t n)
{ size_t i = blockIdx.x * (size_t)blockDim.x + threadIdx.x; if (i < n) p[i] = 0.f; }

__global__ void zero_i32(int* p, size_t n)
{ size_t i = blockIdx.x * (size_t)blockDim.x + threadIdx.x; if (i < n) p[i] = 0; }

__global__ void seg_accum(const int* __restrict__ spx, float* cnt, float* sy, float* sx,
                          int Bn, int H, int W, int S)
{
    size_t i = blockIdx.x * (size_t)blockDim.x + threadIdx.x;
    size_t total = (size_t)Bn * H * W;
    if (i >= total) return;
    int b = (int)(i / ((size_t)H * W));
    int rem = (int)(i % ((size_t)H * W));
    int y = rem / W, x = rem - y * W;
    int s = spx[i] - 1;
    s = s < 0 ? 0 : (s >= S ? S - 1 : s);
    atomicAdd(&cnt[b * S + s], 1.f);
    atomicAdd(&sy[b * S + s], (float)y / (float)H);
    atomicAdd(&sx[b * S + s], (float)x / (float)W);
}

__global__ void seg_info_write(const int* __restrict__ spx,
                               const float* __restrict__ cnt,
                               const float* __restrict__ sy,
                               const float* __restrict__ sx,
                               float* __restrict__ feat,
                               int Bn, int Cfeat, int chOff, int H, int W, int S)
{
    size_t i = blockIdx.x * (size_t)blockDim.x + threadIdx.x;
    size_t total = (size_t)Bn * H * W;
    if (i >= total) return;
    int b = (int)(i / ((size_t)H * W));
    int rem = (int)(i % ((size_t)H * W));
    int s = spx[i] - 1;
    s = s < 0 ? 0 : (s >= S ? S - 1 : s);
    float c = cnt[b * S + s];
    float d = fmaxf(c, 1.f);
    size_t hw = (size_t)H * W;
    size_t base = ((size_t)b * Cfeat + chOff) * hw + rem;
    feat[base]          = c / (float)hw;
    feat[base + hw]     = sy[b * S + s] / d;
    feat[base + 2 * hw] = sx[b * S + s] / d;
}

__global__ void spx_downsample(const int* __restrict__ spx, int* __restrict__ sm,
                               int Bn, int H, int W, int h, int w)
{
    size_t i = blockIdx.x * (size_t)blockDim.x + threadIdx.x;
    size_t total = (size_t)Bn * h * w;
    if (i >= total) return;
    int b = (int)(i / ((size_t)h * w));
    int rem = (int)(i % ((size_t)h * w));
    int y = rem / w, x = rem - y * w;
    int yi = y * H / h, xi = x * W / w;
    sm[i] = spx[((size_t)b * H + yi) * W + xi];
}

__global__ void seg_count_int(const int* __restrict__ spx, int* cnt, int Bn, int HW, int S)
{
    size_t i = blockIdx.x * (size_t)blockDim.x + threadIdx.x;
    size_t total = (size_t)Bn * HW;
    if (i >= total) return;
    int b = (int)(i / HW);
    int s = spx[i] - 1;
    s = s < 0 ? 0 : (s >= S ? S - 1 : s);
    atomicAdd(&cnt[b * S + s], 1);
}

// exclusive prefix sum over S=1024 segments per batch (LDS Hillis-Steele)
__global__ void seg_scan(const int* __restrict__ cnt, int* __restrict__ off, int S)
{
    __shared__ int sh[1024];
    int b = blockIdx.x, t = threadIdx.x;
    int v = cnt[b * S + t];
    sh[t] = v;
    __syncthreads();
    for (int d = 1; d < S; d <<= 1) {
        int add = (t >= d) ? sh[t - d] : 0;
        __syncthreads();
        sh[t] += add;
        __syncthreads();
    }
    off[b * S + t] = sh[t] - v;
}

__global__ void seg_scatter(const int* __restrict__ spx, const int* __restrict__ off,
                            int* __restrict__ cursor, int* __restrict__ order,
                            int Bn, int HW, int S)
{
    size_t i = blockIdx.x * (size_t)blockDim.x + threadIdx.x;
    size_t total = (size_t)Bn * HW;
    if (i >= total) return;
    int b = (int)(i / HW);
    int p = (int)(i % HW);
    int s = spx[i] - 1;
    s = s < 0 ? 0 : (s >= S ? S - 1 : s);
    int pos = off[b * S + s] + atomicAdd(&cursor[b * S + s], 1);
    order[(size_t)b * HW + pos] = p;
}

__device__ __forceinline__ uint32_t mix32(uint32_t h)
{
    h ^= h >> 16; h *= 0x7FEB352Du;
    h ^= h >> 15; h *= 0x846CA68Bu;
    h ^= h >> 16; return h & 0x7FFFFFFFu;
}

// fused: per-superpixel random sampling + first 1D conv/BN/ReLU stage
__global__ void sample_sfconv1(const float* __restrict__ feat,   // [B][C][HW]
                               const int* __restrict__ cnt,
                               const int* __restrict__ off,
                               const int* __restrict__ order,
                               const float* __restrict__ w3, const float* __restrict__ bc,
                               const float* __restrict__ g,  const float* __restrict__ bb,
                               const float* __restrict__ m,  const float* __restrict__ v,
                               float* __restrict__ out1,
                               int Bn, int HW, int S, int C, int J3)
{
    size_t i = blockIdx.x * (size_t)blockDim.x + threadIdx.x;
    size_t total = (size_t)Bn * S * J3;
    if (i >= total) return;
    int j3 = (int)(i % J3);
    int s  = (int)((i / J3) % S);
    int b  = (int)(i / ((size_t)J3 * S));
    int c  = cnt[b * S + s];
    int o  = off[b * S + s];
    int cm = c > 0 ? c : 1;
    int pix[3];
    #pragma unroll
    for (int t = 0; t < 3; ++t) {
        uint32_t j = (uint32_t)(j3 * 3 + t);
        uint32_t r = mix32((uint32_t)b * 0x9E3779B9u ^ (uint32_t)s * 0x85EBCA6Bu ^ j * 0xC2B2AE35u);
        pix[t] = order[(size_t)b * HW + o + (int)(r % (uint32_t)cm)];
    }
    float w0 = w3[0], w1 = w3[1], w2 = w3[2];
    float scale = g[0] * rsqrtf(v[0] + EPS_F);
    float bcv = bc[0], mv = m[0], bbv = bb[0];
    float* orow = out1 + ((size_t)b * S + s) * (size_t)(C * J3);
    for (int ch = 0; ch < C; ++ch) {
        const float* fb = feat + ((size_t)b * C + ch) * HW;
        float val = fb[pix[0]] * w0 + fb[pix[1]] * w1 + fb[pix[2]] * w2 + bcv;
        val = (val - mv) * scale + bbv;
        orow[(size_t)ch * J3 + j3] = fmaxf(val, 0.f);
    }
}

// generic 1D stage: rows of length Lout*3 -> Lout, triple dot + BN + ReLU
__global__ void sfconv1d_stage(const float* __restrict__ in, float* __restrict__ out,
                               const float* __restrict__ w3, const float* __restrict__ bc,
                               const float* __restrict__ g,  const float* __restrict__ bb,
                               const float* __restrict__ m,  const float* __restrict__ v,
                               size_t total, int Lout)
{
    size_t i = blockIdx.x * (size_t)blockDim.x + threadIdx.x;
    if (i >= total) return;
    size_t row = i / Lout;
    int j = (int)(i % Lout);
    const float* r = in + row * (size_t)Lout * 3 + (size_t)j * 3;
    float val = r[0] * w3[0] + r[1] * w3[1] + r[2] * w3[2] + bc[0];
    val = (val - m[0]) * (g[0] * rsqrtf(v[0] + EPS_F)) + bb[0];
    out[i] = fmaxf(val, 0.f);
}

// final: mask by (cnt>0), 1x1 2->1 conv + BN + ReLU
__global__ void sf2d_final(const float* __restrict__ big, const float* __restrict__ small,
                           const int* __restrict__ cb, const int* __restrict__ cs,
                           const float* __restrict__ w, const float* __restrict__ bc,
                           const float* __restrict__ g, const float* __restrict__ bb,
                           const float* __restrict__ m, const float* __restrict__ v,
                           float* __restrict__ out, int Bn, int S, int L)
{
    size_t i = blockIdx.x * (size_t)blockDim.x + threadIdx.x;
    size_t total = (size_t)Bn * S * L;
    if (i >= total) return;
    int s = (int)((i / L) % S);
    int b = (int)(i / ((size_t)L * S));
    float xb = big[i]   * (cb[b * S + s] > 0 ? 1.f : 0.f);
    float xs = small[i] * (cs[b * S + s] > 0 ? 1.f : 0.f);
    float val = w[0] * xb + w[1] * xs + bc[0];
    val = (val - m[0]) * (g[0] * rsqrtf(v[0] + EPS_F)) + bb[0];
    out[i] = fmaxf(val, 0.f);
}

// ---------------------------------------------------------------------------
// Host-side helpers
// ---------------------------------------------------------------------------
struct BnP { const float *g, *b, *m, *v; };

static void launch_conv(hipStream_t st, int Bn,
                        const float* in, int Ci, int IH, int IW,
                        const float* w, const BnP& bn, const float* cbias,
                        int Co, int R, int Sk, int stride, int pad,
                        const float* res, float* out, int CoTotal, int doRelu,
                        __bf16* wpack, float* biasf)
{
    int K = Ci * R * Sk;
    int Kpad = (K + 31) & ~31;
    int CoPad = (Co + 63) & ~63;
    int OH = (IH + 2 * pad - R) / stride + 1;
    int OW = (IW + 2 * pad - Sk) / stride + 1;
    int np = CoPad * Kpad;
    pack_conv_weights<<<(np + 255) / 256, 256, 0, st>>>(
        w, bn.g, bn.b, bn.m, bn.v, cbias, wpack, biasf, Co, CoPad, K, Kpad);
    dim3 grid((OH * OW + 63) / 64, CoPad / 64, Bn);
    if (R == 3 && Sk == 3)
        conv_bn_relu_wmma<3, 3><<<grid, 256, 0, st>>>(
            in, wpack, biasf, res, out, Ci, IH, IW, Co, OH, OW,
            stride, pad, K, Kpad, CoTotal, doRelu);
    else if (R == 1 && Sk == 1)
        conv_bn_relu_wmma<1, 1><<<grid, 256, 0, st>>>(
            in, wpack, biasf, res, out, Ci, IH, IW, Co, OH, OW,
            stride, pad, K, Kpad, CoTotal, doRelu);
    else
        conv_bn_relu_wmma<7, 7><<<grid, 256, 0, st>>>(
            in, wpack, biasf, res, out, Ci, IH, IW, Co, OH, OW,
            stride, pad, K, Kpad, CoTotal, doRelu);
}

#define GRID1(n) dim3((unsigned)(((n) + 255) / 256)), dim3(256)

extern "C" void kernel_launch(void* const* d_in, const int* in_sizes, int n_in,
                              void* d_out, int out_size, void* d_ws, size_t ws_size,
                              hipStream_t stream)
{
    (void)in_sizes; (void)n_in; (void)out_size; (void)ws_size;
    const int Bn = 2, Himg = 384, Wimg = 384, S = 1024;
    const int H2 = 192, W2 = 192, H4 = 96, W4 = 96;
    const int HWb = Himg * Wimg, HWs = H4 * W4;
    const int C32 = 32, J3 = 270;          // 810 samples / 3

    const float* img = (const float*)d_in[0];
    const int*   spx = (const int*)d_in[1];

    // -- walk params pytree in insertion order --
    int pi = 2;
    auto NF = [&]() { return (const float*)d_in[pi++]; };
    auto NB = [&]() { BnP r; r.g = NF(); r.b = NF(); r.m = NF(); r.v = NF(); return r; };

    const float* conv1_w = NF(); BnP conv1_bn = NB();
    struct Blk { const float* w1; BnP bn1; const float* w2; BnP bn2;
                 const float* ws; BnP bns; };
    Blk L[4][2];
    for (int li = 0; li < 4; ++li)
        for (int bi = 0; bi < 2; ++bi) {
            Blk& k = L[li][bi];
            k.w1 = NF(); k.bn1 = NB(); k.w2 = NF(); k.bn2 = NB();
            if ((li == 1 || li == 2) && bi == 0) { k.ws = NF(); k.bns = NB(); }
            else k.ws = nullptr;
        }
    const float* post_w = NF(); const float* post_b = NF(); BnP post_bn = NB();
    const float* dec_w  = NF(); const float* dec_b  = NF(); BnP dec_bn  = NB();
    const float* c1_w   = NF(); const float* c1_b   = NF(); BnP c1_bn   = NB();
    const float* sf2_w  = NF(); const float* sf2_b  = NF(); BnP sf2_bn  = NB();
    struct C1d { const float *w, *b, *g, *bb, *m, *v; } c1d[3];
    for (int i = 0; i < 3; ++i) {
        c1d[i].w = NF(); c1d[i].b = NF(); c1d[i].g = NF();
        c1d[i].bb = NF(); c1d[i].m = NF(); c1d[i].v = NF();
    }

    // -- workspace carve (bump allocator) --
    char* cur = (char*)d_ws;
    auto WS = [&](size_t bytes) { char* p = cur; cur += (bytes + 255) & ~(size_t)255; return p; };
    const size_t featMax = (size_t)Bn * 256 * H4 * W4;   // == Bn*64*H2*W2
    float* bufA   = (float*)WS(featMax * 4);
    float* bufB   = (float*)WS(featMax * 4);
    float* bufC   = (float*)WS(featMax * 4);
    float* outL1  = (float*)WS((size_t)Bn * 64 * H2 * W2 * 4);
    float* feat67 = (float*)WS((size_t)Bn * 67 * HWb * 4);
    float* sft67  = (float*)WS((size_t)Bn * 67 * HWs * 4);
    float* f32b   = (float*)WS((size_t)Bn * C32 * HWb * 4);
    float* f32s   = (float*)WS((size_t)Bn * C32 * HWs * 4);
    __bf16* wpack = (__bf16*)WS((size_t)256 * 2304 * 2);
    float* biasf  = (float*)WS(256 * 4);
    float* segFb  = (float*)WS((size_t)3 * Bn * S * 4);
    float* segFs  = (float*)WS((size_t)3 * Bn * S * 4);
    int* cntIb = (int*)WS((size_t)Bn * S * 4);
    int* offIb = (int*)WS((size_t)Bn * S * 4);
    int* curIb = (int*)WS((size_t)Bn * S * 4);
    int* cntIs = (int*)WS((size_t)Bn * S * 4);
    int* offIs = (int*)WS((size_t)Bn * S * 4);
    int* curIs = (int*)WS((size_t)Bn * S * 4);
    int* ordb  = (int*)WS((size_t)Bn * HWb * 4);
    int* ords  = (int*)WS((size_t)Bn * HWs * 4);
    int* sspx  = (int*)WS((size_t)Bn * HWs * 4);
    float* t1  = (float*)WS((size_t)Bn * S * 8640 * 4);
    float* t2  = (float*)WS((size_t)Bn * S * 2880 * 4);
    float* o3b = (float*)WS((size_t)Bn * S * 960 * 4);
    float* o3s = (float*)WS((size_t)Bn * S * 960 * 4);

    // ================= backbone =================
    launch_conv(stream, Bn, img, 3, Himg, Wimg, conv1_w, conv1_bn, nullptr,
                64, 7, 7, 2, 3, nullptr, bufA, 64, 1, wpack, biasf);
    float* x = bufA;
    // layer1 (64->64, stride 1,1)
    launch_conv(stream, Bn, x, 64, H2, W2, L[0][0].w1, L[0][0].bn1, nullptr,
                64, 3, 3, 1, 1, nullptr, bufB, 64, 1, wpack, biasf);
    launch_conv(stream, Bn, bufB, 64, H2, W2, L[0][0].w2, L[0][0].bn2, nullptr,
                64, 3, 3, 1, 1, x, bufC, 64, 1, wpack, biasf);
    x = bufC;
    launch_conv(stream, Bn, x, 64, H2, W2, L[0][1].w1, L[0][1].bn1, nullptr,
                64, 3, 3, 1, 1, nullptr, bufB, 64, 1, wpack, biasf);
    launch_conv(stream, Bn, bufB, 64, H2, W2, L[0][1].w2, L[0][1].bn2, nullptr,
                64, 3, 3, 1, 1, x, outL1, 64, 1, wpack, biasf);
    x = outL1;
    // layer2 (64->128, strides 2,1)
    launch_conv(stream, Bn, x, 64, H2, W2, L[1][0].w1, L[1][0].bn1, nullptr,
                128, 3, 3, 2, 1, nullptr, bufA, 128, 1, wpack, biasf);
    launch_conv(stream, Bn, x, 64, H2, W2, L[1][0].ws, L[1][0].bns, nullptr,
                128, 1, 1, 2, 0, nullptr, bufC, 128, 0, wpack, biasf);
    launch_conv(stream, Bn, bufA, 128, H4, W4, L[1][0].w2, L[1][0].bn2, nullptr,
                128, 3, 3, 1, 1, bufC, bufB, 128, 1, wpack, biasf);
    x = bufB;
    launch_conv(stream, Bn, x, 128, H4, W4, L[1][1].w1, L[1][1].bn1, nullptr,
                128, 3, 3, 1, 1, nullptr, bufA, 128, 1, wpack, biasf);
    launch_conv(stream, Bn, bufA, 128, H4, W4, L[1][1].w2, L[1][1].bn2, nullptr,
                128, 3, 3, 1, 1, x, bufC, 128, 1, wpack, biasf);
    x = bufC;
    // layer3 (128->256, strides 1,1, shortcut conv on block0)
    launch_conv(stream, Bn, x, 128, H4, W4, L[2][0].w1, L[2][0].bn1, nullptr,
                256, 3, 3, 1, 1, nullptr, bufA, 256, 1, wpack, biasf);
    launch_conv(stream, Bn, x, 128, H4, W4, L[2][0].ws, L[2][0].bns, nullptr,
                256, 1, 1, 1, 0, nullptr, bufB, 256, 0, wpack, biasf);
    launch_conv(stream, Bn, bufA, 256, H4, W4, L[2][0].w2, L[2][0].bn2, nullptr,
                256, 3, 3, 1, 1, bufB, bufC, 256, 1, wpack, biasf);
    x = bufC;
    launch_conv(stream, Bn, x, 256, H4, W4, L[2][1].w1, L[2][1].bn1, nullptr,
                256, 3, 3, 1, 1, nullptr, bufA, 256, 1, wpack, biasf);
    launch_conv(stream, Bn, bufA, 256, H4, W4, L[2][1].w2, L[2][1].bn2, nullptr,
                256, 3, 3, 1, 1, x, bufB, 256, 1, wpack, biasf);
    x = bufB;
    // layer4 (256->256, strides 1,1)
    launch_conv(stream, Bn, x, 256, H4, W4, L[3][0].w1, L[3][0].bn1, nullptr,
                256, 3, 3, 1, 1, nullptr, bufA, 256, 1, wpack, biasf);
    launch_conv(stream, Bn, bufA, 256, H4, W4, L[3][0].w2, L[3][0].bn2, nullptr,
                256, 3, 3, 1, 1, x, bufC, 256, 1, wpack, biasf);
    x = bufC;
    launch_conv(stream, Bn, x, 256, H4, W4, L[3][1].w1, L[3][1].bn1, nullptr,
                256, 3, 3, 1, 1, nullptr, bufA, 256, 1, wpack, biasf);
    launch_conv(stream, Bn, bufA, 256, H4, W4, L[3][1].w2, L[3][1].bn2, nullptr,
                256, 3, 3, 1, 1, x, bufB, 256, 1, wpack, biasf);
    x = bufB;
    // post conv: 256->64 into channels 0..63 of small feature (67-ch stride)
    launch_conv(stream, Bn, x, 256, H4, W4, post_w, post_bn, post_b,
                64, 3, 3, 1, 1, nullptr, sft67, 67, 1, wpack, biasf);

    // deconv: out_L1 -> channels 0..63 of big feature (67-ch stride)
    {
        size_t n = (size_t)Bn * 64 * HWb;
        deconv_bn_relu<<<GRID1(n), 0, stream>>>(
            outL1, dec_w, dec_b, dec_bn.g, dec_bn.b, dec_bn.m, dec_bn.v,
            feat67, Bn, H2, W2, Himg, Wimg, 67);
    }

    // ================= superpixel info maps =================
    zero_f32<<<GRID1(3 * Bn * S), 0, stream>>>(segFb, (size_t)3 * Bn * S);
    seg_accum<<<GRID1((size_t)Bn * HWb), 0, stream>>>(
        spx, segFb, segFb + Bn * S, segFb + 2 * Bn * S, Bn, Himg, Wimg, S);
    seg_info_write<<<GRID1((size_t)Bn * HWb), 0, stream>>>(
        spx, segFb, segFb + Bn * S, segFb + 2 * Bn * S, feat67, Bn, 67, 64, Himg, Wimg, S);
    launch_conv(stream, Bn, feat67, 67, Himg, Wimg, c1_w, c1_bn, c1_b,
                C32, 1, 1, 1, 0, nullptr, f32b, C32, 1, wpack, biasf);

    spx_downsample<<<GRID1((size_t)Bn * HWs), 0, stream>>>(spx, sspx, Bn, Himg, Wimg, H4, W4);
    zero_f32<<<GRID1(3 * Bn * S), 0, stream>>>(segFs, (size_t)3 * Bn * S);
    seg_accum<<<GRID1((size_t)Bn * HWs), 0, stream>>>(
        sspx, segFs, segFs + Bn * S, segFs + 2 * Bn * S, Bn, H4, W4, S);
    seg_info_write<<<GRID1((size_t)Bn * HWs), 0, stream>>>(
        sspx, segFs, segFs + Bn * S, segFs + 2 * Bn * S, sft67, Bn, 67, 64, H4, W4, S);
    launch_conv(stream, Bn, sft67, 67, H4, W4, c1_w, c1_bn, c1_b,
                C32, 1, 1, 1, 0, nullptr, f32s, C32, 1, wpack, biasf);

    // ================= sampling + 1D stack (big) =================
    zero_i32<<<GRID1(Bn * S), 0, stream>>>(cntIb, (size_t)Bn * S);
    zero_i32<<<GRID1(Bn * S), 0, stream>>>(curIb, (size_t)Bn * S);
    seg_count_int<<<GRID1((size_t)Bn * HWb), 0, stream>>>(spx, cntIb, Bn, HWb, S);
    seg_scan<<<dim3(Bn), dim3(S), 0, stream>>>(cntIb, offIb, S);
    seg_scatter<<<GRID1((size_t)Bn * HWb), 0, stream>>>(spx, offIb, curIb, ordb, Bn, HWb, S);
    sample_sfconv1<<<GRID1((size_t)Bn * S * J3), 0, stream>>>(
        f32b, cntIb, offIb, ordb,
        c1d[0].w, c1d[0].b, c1d[0].g, c1d[0].bb, c1d[0].m, c1d[0].v,
        t1, Bn, HWb, S, C32, J3);
    sfconv1d_stage<<<GRID1((size_t)Bn * S * 2880), 0, stream>>>(
        t1, t2, c1d[1].w, c1d[1].b, c1d[1].g, c1d[1].bb, c1d[1].m, c1d[1].v,
        (size_t)Bn * S * 2880, 2880);
    sfconv1d_stage<<<GRID1((size_t)Bn * S * 960), 0, stream>>>(
        t2, o3b, c1d[2].w, c1d[2].b, c1d[2].g, c1d[2].bb, c1d[2].m, c1d[2].v,
        (size_t)Bn * S * 960, 960);

    // ================= sampling + 1D stack (small) =================
    zero_i32<<<GRID1(Bn * S), 0, stream>>>(cntIs, (size_t)Bn * S);
    zero_i32<<<GRID1(Bn * S), 0, stream>>>(curIs, (size_t)Bn * S);
    seg_count_int<<<GRID1((size_t)Bn * HWs), 0, stream>>>(sspx, cntIs, Bn, HWs, S);
    seg_scan<<<dim3(Bn), dim3(S), 0, stream>>>(cntIs, offIs, S);
    seg_scatter<<<GRID1((size_t)Bn * HWs), 0, stream>>>(sspx, offIs, curIs, ords, Bn, HWs, S);
    sample_sfconv1<<<GRID1((size_t)Bn * S * J3), 0, stream>>>(
        f32s, cntIs, offIs, ords,
        c1d[0].w, c1d[0].b, c1d[0].g, c1d[0].bb, c1d[0].m, c1d[0].v,
        t1, Bn, HWs, S, C32, J3);
    sfconv1d_stage<<<GRID1((size_t)Bn * S * 2880), 0, stream>>>(
        t1, t2, c1d[1].w, c1d[1].b, c1d[1].g, c1d[1].bb, c1d[1].m, c1d[1].v,
        (size_t)Bn * S * 2880, 2880);
    sfconv1d_stage<<<GRID1((size_t)Bn * S * 960), 0, stream>>>(
        t2, o3s, c1d[2].w, c1d[2].b, c1d[2].g, c1d[2].bb, c1d[2].m, c1d[2].v,
        (size_t)Bn * S * 960, 960);

    // ================= final mix =================
    sf2d_final<<<GRID1((size_t)Bn * S * 960), 0, stream>>>(
        o3b, o3s, cntIb, cntIs, sf2_w, sf2_b,
        sf2_bn.g, sf2_bn.b, sf2_bn.m, sf2_bn.v,
        (float*)d_out, Bn, S, 960);
}
